// BiRNN_16329465660239
// MI455X (gfx1250) — matrix operand order repaired and checked by reading, not compiled, not run on hardware
//
#include <hip/hip_runtime.h>
#include <hip/hip_bf16.h>

// ---------------------------------------------------------------------------
// BiRNN for MI455X (gfx1250, wave32).
//   k0: convert Whh/fcW -> bf16 workspace
//   k1: Z[dir,t,m,j] = x @ Wih^T + (bih+bhh)      (parallel WMMA GEMM)
//   k2: per-batch-row serial scan (no cross-WG sync needed!) with fused fc.
// All matrix math: v_wmma_f32_16x16x32_bf16, f32 accumulate.
// ---------------------------------------------------------------------------

typedef __bf16 bf16_t;
typedef __attribute__((ext_vector_type(16))) __bf16 v16bf;
typedef __attribute__((ext_vector_type(8)))  __bf16 v8bf;
typedef __attribute__((ext_vector_type(8)))  float  v8f;

#define NB   64
#define LSEQ 1024
#define HID  512

__device__ __forceinline__ v8f wmma_bf16(v16bf a, v16bf b, v8f c) {
  // 8 args: (neg_a, A, neg_b, B, c_mod, C, reuse_a, reuse_b)
  return __builtin_amdgcn_wmma_f32_16x16x32_bf16(false, a, false, b,
                                                 (short)0, c, false, false);
}

__device__ __forceinline__ v8f vzero8() {
  v8f z;
#pragma unroll
  for (int e = 0; e < 8; ++e) z[e] = 0.0f;
  return z;
}

// A-operand (16x32 bf16) from an f32 row. kb = 32*ks + 16*(lane>=16).
// element e -> K=kb+e   (contiguous k order, matches B)
__device__ __forceinline__ v16bf load_a_f32(const float* __restrict__ row, int kb) {
  v16bf a;
#pragma unroll
  for (int e = 0; e < 8; ++e) a[e] = (bf16_t)row[kb + e];
#pragma unroll
  for (int e = 0; e < 8; ++e) a[e + 8] = (bf16_t)row[kb + 8 + e];
  return a;
}

// A-operand from LDS bf16 (already 16B-aligned chunks -> ds_load_b128 x2)
__device__ __forceinline__ v16bf load_a_lds(const bf16_t* hrow) {
  v8bf lo = *(const v8bf*)(hrow);
  v8bf hi = *(const v8bf*)(hrow + 8);
  return __builtin_shufflevector(lo, hi, 0, 1, 2, 3, 4, 5, 6, 7,
                                 8, 9, 10, 11, 12, 13, 14, 15);
}

// ---------------------------------------------------------------------------
// k0: convert recurrent + fc weights to bf16 (read 1024x each in the scan)
// ---------------------------------------------------------------------------
__global__ void __launch_bounds__(256) convert_weights_kernel(
    const float* __restrict__ fwWhh, const float* __restrict__ bwWhh,
    const float* __restrict__ fwFc,  const float* __restrict__ bwFc,
    bf16_t* __restrict__ whh_bf, bf16_t* __restrict__ fcw_bf) {
  int idx = blockIdx.x * 256 + threadIdx.x;        // 0 .. 2*512*512-1
  int dir = idx >> 18;
  int off = idx & 262143;
  whh_bf[idx] = (bf16_t)(dir ? bwWhh[off] : fwWhh[off]);
  fcw_bf[idx] = (bf16_t)(dir ? bwFc[off]  : fwFc[off]);
}

// ---------------------------------------------------------------------------
// k1: Z = x @ Wih^T + (bih+bhh), flat rows r = dir*65536 + t*64 + m.
// Block: 8 waves, 128 rows x 64 cols; Wih col-slice (64x512) staged in LDS.
// Grid: 2 dirs * 1024 rowblocks * 8 colslices = 16384 blocks.
// ---------------------------------------------------------------------------
__global__ void __launch_bounds__(256) zgemm_kernel(
    const float* __restrict__ x,
    const float* __restrict__ fwWih, const float* __restrict__ bwWih,
    const float* __restrict__ fwBih, const float* __restrict__ fwBhh,
    const float* __restrict__ bwBih, const float* __restrict__ bwBhh,
    float* __restrict__ Z) {
  __shared__ bf16_t wlds[64][520];                 // padded rows (bank spread)

  const int bid = blockIdx.x;
  const int cs  = bid & 7;                         // col slice -> j0 = cs*64
  const int rb  = (bid >> 3) & 1023;               // 128-row block
  const int dir = bid >> 13;
  const int j0  = cs << 6;

  const int tid  = threadIdx.x;
  const int lane = tid & 31;
  const int w    = tid >> 5;
  const int nl   = lane & 15;
  const int hi   = lane >> 4;

  const float* Wih = dir ? bwWih : fwWih;
  const float* bih = dir ? bwBih : fwBih;
  const float* bhh = dir ? bwBhh : fwBhh;

  // stage Wih[j0 .. j0+63][0..511] as bf16 in LDS
  for (int i = tid; i < 64 * 512; i += 256) {
    int r = i >> 9, k = i & 511;
    wlds[r][k] = (bf16_t)Wih[(size_t)(j0 + r) * 512 + k];
  }
  __syncthreads();

  // wave -> 16-row tile: t = 2*rb + (w>>2), m0 = (w&3)*16
  const int tt   = rb * 2 + (w >> 2);
  const int m0   = (w & 3) << 4;
  const int tsrc = dir ? (LSEQ - 1 - tt) : tt;     // bw direction reads x flipped
  const float* xrow = x + ((size_t)(m0 + nl) * LSEQ + tsrc) * 512;

  v8f acc[4];
#pragma unroll
  for (int c = 0; c < 4; ++c) acc[c] = vzero8();

  for (int ks = 0; ks < 16; ++ks) {
    v16bf a = load_a_f32(xrow, ks * 32 + hi * 16);
    const int kbB = ks * 32 + hi * 16;
#pragma unroll
    for (int c = 0; c < 4; ++c) {
      v16bf b = *(const v16bf*)(&wlds[c * 16 + nl][kbB]);
      acc[c] = wmma_bf16(a, b, acc[c]);
    }
  }

  const size_t rbase = (size_t)dir * 65536 + (size_t)tt * 64 + m0;
#pragma unroll
  for (int c = 0; c < 4; ++c) {
    const int col = j0 + c * 16 + nl;
    const float bv = bih[col] + bhh[col];
#pragma unroll
    for (int e = 0; e < 8; ++e) {
      const int M = e + hi * 8;                    // C/D layout
      Z[(rbase + M) * 512 + col] = acc[c][e] + bv;
    }
  }
}

// ---------------------------------------------------------------------------
// k2: serial scan, fused fc. 8 blocks = 2 dirs x 4 row-blocks of 16 rows.
// Recurrence is independent per batch row -> zero cross-WG synchronization.
// 32 waves: wave w owns 16 output columns j0 = 16*w. One barrier per step.
// h kept in LDS bf16, double-buffered. Whh/fcW streamed bf16 from L2.
// ---------------------------------------------------------------------------
__global__ void __launch_bounds__(1024) scan_kernel(
    const float* __restrict__ h0,                  // [2][64][512]
    const float* __restrict__ Z,                   // [2][1024][64][512]
    const bf16_t* __restrict__ whh_bf,             // [2][512][512]
    const bf16_t* __restrict__ fcw_bf,             // [2][512][512]
    const float* __restrict__ fwFcb, const float* __restrict__ bwFcb,
    float* __restrict__ y,                         // [64][1024][1024]
    float* __restrict__ hout) {                    // [2][64][512]
  __shared__ bf16_t hbuf[2][16][520];              // double-buffered h (bf16)

  const int dir = blockIdx.x >> 2;
  const int m0  = (blockIdx.x & 3) << 4;

  const int tid  = threadIdx.x;
  const int lane = tid & 31;
  const int w    = tid >> 5;
  const int nl   = lane & 15;
  const int hi   = lane >> 4;
  const int j0   = w << 4;

  const bf16_t* Wd  = whh_bf + (size_t)dir * 262144;
  const bf16_t* Fd  = fcw_bf + (size_t)dir * 262144;
  const float*  fcb = dir ? bwFcb : fwFcb;
  const float*  Zd  = Z + (size_t)dir * LSEQ * NB * 512;

  // seed h from h0
  for (int i = tid; i < 16 * 512; i += 1024) {
    int r = i >> 9, c = i & 511;
    hbuf[0][r][c] = (bf16_t)h0[((size_t)dir * NB + m0 + r) * 512 + c];
  }
  __syncthreads();

  const float fcbv = fcb[j0 + nl];

  for (int t = 0; t < LSEQ; ++t) {
    const int p = t & 1, q = p ^ 1;
    const float* zt = Zd + ((size_t)t * NB + m0) * 512;  // 16x512 tile, contiguous

    if (t + 1 < LSEQ)                                    // prefetch next Z tile
      __builtin_prefetch(zt + (size_t)NB * 512 + tid * 8, 0, 0);

    // h_new = tanh(Z_t + h_prev @ Whh^T)   (this wave's 16 columns)
    v8f acc = vzero8();
#pragma unroll
    for (int ks = 0; ks < 16; ++ks) {
      v16bf a = load_a_lds(&hbuf[p][nl][ks * 32 + hi * 16]);
      v16bf b = *(const v16bf*)(Wd + (size_t)(j0 + nl) * 512 + ks * 32 + hi * 16);
      acc = wmma_bf16(a, b, acc);
    }
#pragma unroll
    for (int e = 0; e < 8; ++e) {
      const int M = e + hi * 8;
      const float v = tanhf(acc[e] + zt[(size_t)M * 512 + j0 + nl]);
      hbuf[q][M][j0 + nl] = (bf16_t)v;
      if (t == LSEQ - 1)
        hout[((size_t)dir * NB + m0 + M) * 512 + j0 + nl] = v;
    }
    __syncthreads();                                 // publish h_new to all waves

    // y_t = h_new @ fcW^T + fcb  -> straight to d_out (fw cols / bw cols)
    v8f ya = vzero8();
#pragma unroll
    for (int ks = 0; ks < 16; ++ks) {
      v16bf a = load_a_lds(&hbuf[q][nl][ks * 32 + hi * 16]);
      v16bf b = *(const v16bf*)(Fd + (size_t)(j0 + nl) * 512 + ks * 32 + hi * 16);
      ya = wmma_bf16(a, b, ya);
    }
#pragma unroll
    for (int e = 0; e < 8; ++e) {
      const int M = e + hi * 8;
      y[((size_t)(m0 + M) * LSEQ + t) * 1024 + (dir << 9) + j0 + nl] = ya[e] + fcbv;
    }
  }
}

// ---------------------------------------------------------------------------
extern "C" void kernel_launch(void* const* d_in, const int* in_sizes, int n_in,
                              void* d_out, int out_size, void* d_ws, size_t ws_size,
                              hipStream_t stream) {
  const float* x     = (const float*)d_in[0];
  const float* h0    = (const float*)d_in[1];
  const float* fwWih = (const float*)d_in[2];
  const float* fwWhh = (const float*)d_in[3];
  const float* fwBih = (const float*)d_in[4];
  const float* fwBhh = (const float*)d_in[5];
  const float* fwFcW = (const float*)d_in[6];
  const float* fwFcb = (const float*)d_in[7];
  const float* bwWih = (const float*)d_in[8];
  const float* bwWhh = (const float*)d_in[9];
  const float* bwBih = (const float*)d_in[10];
  const float* bwBhh = (const float*)d_in[11];
  const float* bwFcW = (const float*)d_in[12];
  const float* bwFcb = (const float*)d_in[13];

  char* ws = (char*)d_ws;
  bf16_t* whh_bf = (bf16_t*)(ws);                      // 1 MB  [2][512][512] bf16
  bf16_t* fcw_bf = (bf16_t*)(ws + (1u << 20));         // 1 MB
  float*  Zbuf   = (float*) (ws + (2u << 20));         // 256 MB [2][1024][64][512] f32

  float* y    = (float*)d_out;
  float* hout = y + (size_t)NB * LSEQ * 1024;

  convert_weights_kernel<<<2048, 256, 0, stream>>>(fwWhh, bwWhh, fwFcW, bwFcW,
                                                   whh_bf, fcw_bf);
  zgemm_kernel<<<16384, 256, 0, stream>>>(x, fwWih, bwWih, fwBih, fwBhh,
                                          bwBih, bwBhh, Zbuf);
  scan_kernel<<<8, 1024, 0, stream>>>(h0, Zbuf, whh_bf, fcw_bf,
                                      fwFcb, bwFcb, y, hout);
}
